// ViTImageTransformer_16741782520159
// MI455X (gfx1250) — compile-verified, matrix operand order checked by reading, not run on hardware
//
#include <hip/hip_runtime.h>

// ---------------------------------------------------------------------------
// Types for CDNA5 WMMA (wave32): v_wmma_f32_16x16x32_bf16
// ---------------------------------------------------------------------------
typedef __attribute__((ext_vector_type(16))) __bf16 bf16x16;
typedef __attribute__((ext_vector_type(8)))  float  floatx8;
typedef int v4i __attribute__((vector_size(16)));

union FragBF {
    bf16x16 v;
    uint4   u[2];
};
static_assert(sizeof(FragBF) == 32, "frag must be 32B");

__device__ __forceinline__ floatx8 zero8() {
    floatx8 z;
#pragma unroll
    for (int i = 0; i < 8; ++i) z[i] = 0.0f;
    return z;
}

// ---------------------------------------------------------------------------
// CDNA5 async global->LDS copy (ASYNCcnt path) with fallbacks.
// ---------------------------------------------------------------------------
#define AS1 __attribute__((address_space(1)))
#define AS3 __attribute__((address_space(3)))

#if __has_builtin(__builtin_amdgcn_global_load_async_to_lds_b128)
#define ASYNC_CP16(g, l)                                                       \
    __builtin_amdgcn_global_load_async_to_lds_b128((AS1 v4i*)(g),              \
                                                   (AS3 v4i*)(l), 0, 0)
#else
#define ASYNC_CP16(g, l)                                                       \
    asm volatile("global_load_async_to_lds_b128 %0, %1, off"                   \
                 :: "v"((unsigned)(size_t)(AS3 void*)(l)),                     \
                    "v"((void*)(g))                                            \
                 : "memory")
#endif

#if __has_builtin(__builtin_amdgcn_s_wait_asynccnt)
#define WAIT_ASYNC(n) __builtin_amdgcn_s_wait_asynccnt(n)
#else
#define WAIT_ASYNC(n) asm volatile("s_wait_asynccnt %0" :: "i"(n) : "memory")
#endif

// ---------------------------------------------------------------------------
// LayerNorm over E=768, f32 in -> bf16 out.  One block (256 thr) per row.
// ---------------------------------------------------------------------------
__global__ __launch_bounds__(256) void ln_bf16_kernel(
    const float* __restrict__ x, const float* __restrict__ g,
    const float* __restrict__ bta, __bf16* __restrict__ out)
{
    const int row = blockIdx.x;
    const int tid = threadIdx.x;
    const float* xr = x + (size_t)row * 768;
    float v0 = xr[tid], v1 = xr[tid + 256], v2 = xr[tid + 512];
    float s  = v0 + v1 + v2;
    float s2 = v0 * v0 + v1 * v1 + v2 * v2;

    __shared__ float rs[256], rq[256];
    rs[tid] = s; rq[tid] = s2;
    __syncthreads();
#pragma unroll
    for (int off = 128; off > 0; off >>= 1) {
        if (tid < off) { rs[tid] += rs[tid + off]; rq[tid] += rq[tid + off]; }
        __syncthreads();
    }
    const float mean = rs[0] * (1.0f / 768.0f);
    const float var  = rq[0] * (1.0f / 768.0f) - mean * mean;
    const float rstd = rsqrtf(var + 1e-6f);

    __bf16* orow = out + (size_t)row * 768;
    orow[tid]       = (__bf16)((v0 - mean) * rstd * g[tid]       + bta[tid]);
    orow[tid + 256] = (__bf16)((v1 - mean) * rstd * g[tid + 256] + bta[tid + 256]);
    orow[tid + 512] = (__bf16)((v2 - mean) * rstd * g[tid + 512] + bta[tid + 512]);
}

// ---------------------------------------------------------------------------
// Weight convert + transpose: in f32 [K,N] row-major -> out bf16 [N,K].
// ---------------------------------------------------------------------------
__global__ __launch_bounds__(256) void convT_kernel(
    const float* __restrict__ in, __bf16* __restrict__ out, int K, int N)
{
    size_t i = (size_t)blockIdx.x * 256 + threadIdx.x;
    if (i >= (size_t)K * N) return;
    int n = (int)(i / K), k = (int)(i % K);
    out[i] = (__bf16)in[(size_t)k * N + n];
}

// ---------------------------------------------------------------------------
// Tiled GEMM: C[M,N] = A[M,K](bf16) @ W (W given transposed bf16 [N,K]).
// Block 256 thr = 8 waves; tile 128x128, BK=32. Each wave: 32x64 = 2x4 WMMA.
// Global->LDS staging uses CDNA5 async-to-LDS copies with a 2-deep pipeline
// (ASYNCcnt completes in order, so s_wait_asynccnt<=4 == "previous tile done").
// Epilogue: +bias, optional exact GELU, write bf16 and/or f32(+residual).
// ---------------------------------------------------------------------------
#define BM 128
#define BN 128
#define BKK 32
#define LDT 40   // halves per LDS row (32 data + 8 pad), 80B = 16B aligned

__global__ __launch_bounds__(256) void gemm_bf16_kernel(
    const __bf16* __restrict__ A, const __bf16* __restrict__ Wt,
    const float* __restrict__ bias, __bf16* __restrict__ outB,
    float* __restrict__ outF, const float* __restrict__ resid,
    int M, int N, int K, int doGelu)
{
    __shared__ __align__(16) __bf16 As[2][BM * LDT];
    __shared__ __align__(16) __bf16 Bs[2][BN * LDT];

    const int tid  = threadIdx.x;
    const int lane = tid & 31;
    const int w    = tid >> 5;
    const int wm   = w & 3;        // 4 waves along M
    const int wn   = w >> 2;       // 2 waves along N
    const int m0   = blockIdx.y * BM;
    const int n0   = blockIdx.x * BN;
    const int sel  = (lane >> 4) & 1;
    const int l15  = lane & 15;

    floatx8 acc[2][4];
#pragma unroll
    for (int i = 0; i < 2; ++i)
#pragma unroll
        for (int j = 0; j < 4; ++j) acc[i][j] = zero8();

    const int lr = tid >> 2;          // 0..63
    const int lc = (tid & 3) * 8;     // half offset within BK

    const __bf16* gA0 = A  + (size_t)(m0 + lr)      * K + lc;
    const __bf16* gA1 = A  + (size_t)(m0 + lr + 64) * K + lc;
    const __bf16* gB0 = Wt + (size_t)(n0 + lr)      * K + lc;
    const __bf16* gB1 = Wt + (size_t)(n0 + lr + 64) * K + lc;

    auto issue = [&](int buf, int k0) {
        ASYNC_CP16(gA0 + k0, &As[buf][lr * LDT + lc]);
        ASYNC_CP16(gA1 + k0, &As[buf][(lr + 64) * LDT + lc]);
        ASYNC_CP16(gB0 + k0, &Bs[buf][lr * LDT + lc]);
        ASYNC_CP16(gB1 + k0, &Bs[buf][(lr + 64) * LDT + lc]);
    };

    const int nk = K / BKK;
    issue(0, 0);

    for (int ki = 0; ki < nk; ++ki) {
        const int cur = ki & 1;
        if (ki + 1 < nk) {
            issue(cur ^ 1, (ki + 1) * BKK);
            WAIT_ASYNC(4);   // 4 newer loads may fly; older (tile ki) complete
        } else {
            WAIT_ASYNC(0);
        }
        __syncthreads();

        FragBF a[2], b[4];
#pragma unroll
        for (int i = 0; i < 2; ++i) {
            const __bf16* p = &As[cur][(wm * 32 + i * 16 + l15) * LDT + sel * 8];
            a[i].u[0] = *(const uint4*)p;
            a[i].u[1] = *(const uint4*)(p + 16);
        }
#pragma unroll
        for (int j = 0; j < 4; ++j) {
            const __bf16* p = &Bs[cur][(wn * 64 + j * 16 + l15) * LDT + sel * 8];
            b[j].u[0] = *(const uint4*)p;
            b[j].u[1] = *(const uint4*)(p + 16);
        }
#pragma unroll
        for (int i = 0; i < 2; ++i)
#pragma unroll
            for (int j = 0; j < 4; ++j)
                acc[i][j] = __builtin_amdgcn_wmma_f32_16x16x32_bf16(
                    false, a[i].v, false, b[j].v, (short)0, acc[i][j], false, false);
        __syncthreads();
    }

    // Epilogue. D layout: VGPR r, lanes 0-15 -> M=r, lanes 16-31 -> M=8+r; N=lane&15.
#pragma unroll
    for (int i = 0; i < 2; ++i)
#pragma unroll
        for (int j = 0; j < 4; ++j)
#pragma unroll
            for (int r = 0; r < 8; ++r) {
                const int m = m0 + wm * 32 + i * 16 + (sel ? 8 + r : r);
                const int n = n0 + wn * 64 + j * 16 + l15;
                float vv = acc[i][j][r] + bias[n];
                if (doGelu) vv = 0.5f * vv * (1.0f + erff(vv * 0.70710678118654752f));
                const size_t idx = (size_t)m * N + n;
                if (outB) outB[idx] = (__bf16)vv;
                if (outF) outF[idx] = resid[idx] + vv;
            }
}

// ---------------------------------------------------------------------------
// Fused attention per (b,h). Block = 128 thr = 4 waves, one 16-row strip each.
//   score = softmax( (Q @ K^T) / sqrt(768) )   (Q,K,V bf16 [B*S, 768] rows)
//   attn  = V @ score ; x = image + attn  (flat [B,H,S,T] layout == reference)
// ---------------------------------------------------------------------------
__global__ __launch_bounds__(128) void attn_kernel(
    const __bf16* __restrict__ q, const __bf16* __restrict__ k,
    const __bf16* __restrict__ v, const float* __restrict__ image,
    float* __restrict__ x)
{
    const int bh = blockIdx.x;           // b*12 + h
    const int b  = bh / 12;
    const int h  = bh % 12;
    const int tid  = threadIdx.x;
    const int lane = tid & 31;
    const int w    = tid >> 5;
    const int sel  = (lane >> 4) & 1;
    const int l15  = lane & 15;

    __shared__ __align__(16) float  sc[64 * 66];
    __shared__ __align__(16) __bf16 scT[64 * 72];

    const size_t rowbase = (size_t)(b * 64) * 768 + h * 64;
    const __bf16* qb = q + rowbase;
    const __bf16* kb = k + rowbase;
    const __bf16* vb = v + rowbase;
    const int srow = w * 16 + l15;

    // ---- phase 1: score = Q @ K^T --------------------------------------
    floatx8 acc[4];
#pragma unroll
    for (int t = 0; t < 4; ++t) acc[t] = zero8();
#pragma unroll
    for (int kc = 0; kc < 2; ++kc) {
        FragBF a;
        const __bf16* ap = qb + (size_t)srow * 768 + kc * 32 + sel * 8;
        a.u[0] = *(const uint4*)ap;
        a.u[1] = *(const uint4*)(ap + 16);
#pragma unroll
        for (int tt = 0; tt < 4; ++tt) {
            FragBF bf;
            const __bf16* bp = kb + (size_t)(tt * 16 + l15) * 768 + kc * 32 + sel * 8;
            bf.u[0] = *(const uint4*)bp;
            bf.u[1] = *(const uint4*)(bp + 16);
            acc[tt] = __builtin_amdgcn_wmma_f32_16x16x32_bf16(
                false, a.v, false, bf.v, (short)0, acc[tt], false, false);
        }
    }
#pragma unroll
    for (int tt = 0; tt < 4; ++tt)
#pragma unroll
        for (int r = 0; r < 8; ++r)
            sc[(w * 16 + (sel ? 8 + r : r)) * 66 + tt * 16 + l15] = acc[tt][r];
    __syncthreads();

    // ---- phase 2: softmax rows, store transposed bf16 ------------------
    if (tid < 64) {
        const int r = tid;
        const float scale = 0.03608439182435161f; // 1/sqrt(768)
        float mx = -3.0e38f;
        for (int t = 0; t < 64; ++t) mx = fmaxf(mx, sc[r * 66 + t]);
        mx *= scale;
        float sum = 0.0f;
        for (int t = 0; t < 64; ++t) sum += __expf(sc[r * 66 + t] * scale - mx);
        const float inv = 1.0f / sum;
        for (int t = 0; t < 64; ++t)
            scT[t * 72 + r] = (__bf16)(__expf(sc[r * 66 + t] * scale - mx) * inv);
    }
    __syncthreads();

    // ---- phase 3: attn = V @ score ;  x = image + attn ------------------
#pragma unroll
    for (int t = 0; t < 4; ++t) acc[t] = zero8();
#pragma unroll
    for (int kc = 0; kc < 2; ++kc) {
        FragBF a;
        const __bf16* ap = vb + (size_t)srow * 768 + kc * 32 + sel * 8;
        a.u[0] = *(const uint4*)ap;
        a.u[1] = *(const uint4*)(ap + 16);
#pragma unroll
        for (int tt = 0; tt < 4; ++tt) {
            FragBF bf;
            const __bf16* bp = scT + (tt * 16 + l15) * 72 + kc * 32 + sel * 8;
            bf.u[0] = *(const uint4*)bp;
            bf.u[1] = *(const uint4*)(bp + 16);
            acc[tt] = __builtin_amdgcn_wmma_f32_16x16x32_bf16(
                false, a.v, false, bf.v, (short)0, acc[tt], false, false);
        }
    }
#pragma unroll
    for (int tt = 0; tt < 4; ++tt)
#pragma unroll
        for (int r = 0; r < 8; ++r) {
            const int s_ = w * 16 + (sel ? 8 + r : r);
            const int t_ = tt * 16 + l15;
            const size_t idx = (size_t)bh * 4096 + (size_t)s_ * 64 + t_;
            x[idx] = image[idx] + acc[tt][r];
        }
}

// ---------------------------------------------------------------------------
// Host launcher
// ---------------------------------------------------------------------------
extern "C" void kernel_launch(void* const* d_in, const int* in_sizes, int n_in,
                              void* d_out, int out_size, void* d_ws, size_t ws_size,
                              hipStream_t stream)
{
    (void)in_sizes; (void)n_in; (void)out_size; (void)ws_size;
    const float* image = (const float*)d_in[0];
    const float* ln_g  = (const float*)d_in[1];
    const float* ln_b  = (const float*)d_in[2];
    const float* Wq = (const float*)d_in[3];
    const float* bq = (const float*)d_in[4];
    const float* Wk = (const float*)d_in[5];
    const float* bk = (const float*)d_in[6];
    const float* Wv = (const float*)d_in[7];
    const float* bv = (const float*)d_in[8];
    const float* W1 = (const float*)d_in[9];
    const float* b1 = (const float*)d_in[10];
    const float* W2 = (const float*)d_in[11];
    const float* b2 = (const float*)d_in[12];
    float* out = (float*)d_out;

    const size_t ROWS = 65536;   // B*S
    char* ws = (char*)d_ws;
    size_t off = 0;
    auto take = [&](size_t bytes) {
        size_t o = off; off += (bytes + 255) & ~(size_t)255; return o;
    };
    __bf16* hn  = (__bf16*)(ws + take(ROWS * 768  * 2));
    __bf16* qb  = (__bf16*)(ws + take(ROWS * 768  * 2));
    __bf16* kbf = (__bf16*)(ws + take(ROWS * 768  * 2));
    __bf16* vbf = (__bf16*)(ws + take(ROWS * 768  * 2));
    float*  x   = (float*) (ws + take(ROWS * 768  * 4));
    __bf16* act = (__bf16*)(ws + take(ROWS * 3072 * 2));
    __bf16* Wqt = (__bf16*)(ws + take((size_t)768  * 768  * 2));
    __bf16* Wkt = (__bf16*)(ws + take((size_t)768  * 768  * 2));
    __bf16* Wvt = (__bf16*)(ws + take((size_t)768  * 768  * 2));
    __bf16* W1t = (__bf16*)(ws + take((size_t)768  * 3072 * 2));
    __bf16* W2t = (__bf16*)(ws + take((size_t)3072 * 768  * 2));

    // 1) weights -> bf16 transposed [N,K]
    convT_kernel<<<(768 * 768  + 255) / 256, 256, 0, stream>>>(Wq, Wqt, 768, 768);
    convT_kernel<<<(768 * 768  + 255) / 256, 256, 0, stream>>>(Wk, Wkt, 768, 768);
    convT_kernel<<<(768 * 768  + 255) / 256, 256, 0, stream>>>(Wv, Wvt, 768, 768);
    convT_kernel<<<(768 * 3072 + 255) / 256, 256, 0, stream>>>(W1, W1t, 768, 3072);
    convT_kernel<<<(3072 * 768 + 255) / 256, 256, 0, stream>>>(W2, W2t, 3072, 768);

    // 2) LN(image) -> hn (bf16)
    ln_bf16_kernel<<<(int)ROWS, 256, 0, stream>>>(image, ln_g, ln_b, hn);

    // 3) Q/K/V projections (bf16 WMMA, f32 accum, bf16 out)
    dim3 gqkv(768 / BN, (int)(ROWS / BM));
    gemm_bf16_kernel<<<gqkv, 256, 0, stream>>>(hn, Wqt, bq, qb,  nullptr, nullptr,
                                               (int)ROWS, 768, 768, 0);
    gemm_bf16_kernel<<<gqkv, 256, 0, stream>>>(hn, Wkt, bk, kbf, nullptr, nullptr,
                                               (int)ROWS, 768, 768, 0);
    gemm_bf16_kernel<<<gqkv, 256, 0, stream>>>(hn, Wvt, bv, vbf, nullptr, nullptr,
                                               (int)ROWS, 768, 768, 0);

    // 4) attention + residual -> x (f32)
    attn_kernel<<<1024 * 12, 128, 0, stream>>>(qb, kbf, vbf, image, x);

    // 5) LN(x) -> hn (reuse)
    ln_bf16_kernel<<<(int)ROWS, 256, 0, stream>>>(x, ln_g, ln_b, hn);

    // 6) MLP up + GELU -> act (bf16)
    dim3 gm1(3072 / BN, (int)(ROWS / BM));
    gemm_bf16_kernel<<<gm1, 256, 0, stream>>>(hn, W1t, b1, act, nullptr, nullptr,
                                              (int)ROWS, 3072, 768, 1);

    // 7) MLP down + bias + residual(x) -> out (f32)
    dim3 gm2(768 / BN, (int)(ROWS / BM));
    gemm_bf16_kernel<<<gm2, 256, 0, stream>>>(act, W2t, b2, nullptr, out, x,
                                              (int)ROWS, 768, 3072, 0);
}
